// Model_39676907882990
// MI455X (gfx1250) — matrix operand order repaired and checked by reading, not compiled, not run on hardware
//
#include <hip/hip_runtime.h>
#include <hip/hip_bf16.h>
#include <stdint.h>

// ---------------------------------------------------------------------------
// B=16, S=1024, D=1024
// scores = (x @ key_w^T) * 32 ; softmax over k ; dropout p=0.1 ; value_w @ attn
//
// bf16 WMMA pipeline, fp32 accumulate:
//   1) cvt_bf16 : x, key_w, value_w fp32 -> bf16 (bandwidth pass)
//   2) qk_softmax_dropout : 64-row x 1024-col blocks (softmax needs the full
//      k row in-block, so k cannot be tiled; instead we widen M). 8 waves x
//      (64x128) tiles: each B tile feeds 4 WMMAs -> key_w L2 traffic falls to
//      512 MB (vs 2 GB at 16 rows), 0.75 loads/wmma. Fused softmax
//      (cross-wave LDS reduce) + dropout; stores attn TRANSPOSED
//      (attnT[b][k][j]) -- the WMMA C layout makes the transpose free.
//   3) av_gemm : 128x128 2-D blocks (no softmax constraint) so attnT/value_w
//      tiles are shared by all 8 waves through WGP$ -> ~256 MB L2 traffic.
//      All intermediates (68 MB) stay resident in the 192 MB L2.
//
// Workspace: (B*S*D + 2*D*D + B*S*S)*2 = 68 MiB.
// Dropout RNG: deterministic counter hash (JAX threefry not reproduced).
// ---------------------------------------------------------------------------

constexpr int B = 16, S = 1024, D = 1024;
constexpr float SCORE_SCALE = 32.0f;          // divide by D**-0.5
constexpr float KEEP_P  = 0.9f;
constexpr float INV_KEEP = 1.0f / 0.9f;

typedef __attribute__((ext_vector_type(16))) __bf16 v16bf;
typedef __attribute__((ext_vector_type(8)))  float  v8f;

union ABu { uint4 u[2]; v16bf v; };

__device__ __forceinline__ unsigned short f32_to_bf16(float f) {
  union { float f; unsigned int u; } c; c.f = f;
  unsigned int r = c.u + 0x7FFFu + ((c.u >> 16) & 1u);   // round-nearest-even
  return (unsigned short)(r >> 16);
}

__device__ __forceinline__ float hash_u01(unsigned int x) {
  x ^= x >> 17; x *= 0xed5ad4bbu;
  x ^= x >> 11; x *= 0xac4c1b51u;
  x ^= x >> 15; x *= 0x31848babu;
  x ^= x >> 14;
  return (float)(x >> 8) * (1.0f / 16777216.0f);
}

// ------------------------- fp32 -> bf16 conversion -------------------------
__global__ void __launch_bounds__(256) cvt_bf16(const float* __restrict__ src,
                                                unsigned short* __restrict__ dst,
                                                int n4) {
  int i = blockIdx.x * blockDim.x + threadIdx.x;
  if (i >= n4) return;
  float4 f = ((const float4*)src)[i];
  uint2 p;
  p.x = (unsigned)f32_to_bf16(f.x) | ((unsigned)f32_to_bf16(f.y) << 16);
  p.y = (unsigned)f32_to_bf16(f.z) | ((unsigned)f32_to_bf16(f.w) << 16);
  ((uint2*)dst)[i] = p;
}

// ---------------- GEMM1 + softmax + dropout, transposed store --------------
// grid: B * S/64 blocks; block: 256 threads = 8 waves.
// Block owns 64 rows (j) of batch b; wave w owns 128 columns (k),
// computing four 16x128 M-subtiles that all share every B tile.
__global__ void __launch_bounds__(256) qk_softmax_dropout(
    const unsigned short* __restrict__ xb,     // [B,S,D]  bf16
    const unsigned short* __restrict__ kb,     // [D,D]    bf16 (key_w, row=k)
    unsigned short* __restrict__ attnT) {      // [B,S(k),S(j)] bf16
  __shared__ float redbuf[8][64];
  __shared__ float rowstat[64];

  const int wave = threadIdx.x >> 5;
  const int lane = threadIdx.x & 31;
  const int hl   = lane >> 4;      // half-wave: selects K-chunk per ISA layout
  const int l15  = lane & 15;
  const int b    = blockIdx.x >> 4;          // 16 row-tiles per batch
  const int j0   = (blockIdx.x & 15) << 6;   // 64-row tile
  const int wk0  = wave << 7;                // 128 k-columns per wave

  const v8f vzero = {0.f,0.f,0.f,0.f,0.f,0.f,0.f,0.f};
  v8f acc[4][8];
#pragma unroll
  for (int r = 0; r < 4; ++r)
#pragma unroll
    for (int t = 0; t < 8; ++t) acc[r][t] = vzero;

  // A operand (16x32 bf16): lane(l15,hl) holds row j0+r*16+l15,
  //   K = d + hl*8 + {0..7} and + {16..23}  -> two b128 chunks per subtile.
  const uint4* Ab[4];
#pragma unroll
  for (int r = 0; r < 4; ++r)
    Ab[r] = (const uint4*)(xb + (size_t)(b * S + j0 + (r << 4) + l15) * D);

  for (int d = 0; d < D; d += 32) {
    const int c0 = (d >> 3) + hl;
    ABu A[4];
#pragma unroll
    for (int r = 0; r < 4; ++r) {
      A[r].u[0] = Ab[r][c0];
      A[r].u[1] = Ab[r][c0 + 2];
    }
    __builtin_prefetch(Ab[0] + c0 + 4, 0, 1);  // next K-tile -> global_prefetch
    __builtin_prefetch(Ab[2] + c0 + 4, 0, 1);
#pragma unroll
    for (int t = 0; t < 8; ++t) {
      // B operand (32x16 bf16): lane holds column N=l15 (k = wk0+16t+l15),
      //   K = d + hl*8 + {0..7} and + {16..23} -> key_w row: two b128 loads,
      //   shared by all four M-subtiles.
      const uint4* Bp = (const uint4*)(kb + (size_t)(wk0 + (t << 4) + l15) * D
                                          + d + (hl << 3));
      ABu Bt;
      Bt.u[0] = Bp[0];
      Bt.u[1] = Bp[2];
#pragma unroll
      for (int r = 0; r < 4; ++r)
        acc[r][t] = __builtin_amdgcn_wmma_f32_16x16x32_bf16(
            false, A[r].v, false, Bt.v, (short)0, acc[r][t], false, false);
    }
  }

  // C layout: VGPR v, lanes 0-15 -> m=v, lanes 16-31 -> m=v+8; col n=l15.
  // Global row of acc[r][t][v] is j0 + r*16 + hl*8 + v.
  // --- row max (scale by 32 first; monotonic) ---
  float pmax[4][8];
#pragma unroll
  for (int r = 0; r < 4; ++r)
#pragma unroll
    for (int v = 0; v < 8; ++v) {
      float m = -3.0e38f;
#pragma unroll
      for (int t = 0; t < 8; ++t) {
        acc[r][t][v] *= SCORE_SCALE;
        m = fmaxf(m, acc[r][t][v]);
      }
      pmax[r][v] = m;
    }
  for (int off = 8; off; off >>= 1)
#pragma unroll
    for (int r = 0; r < 4; ++r)
#pragma unroll
      for (int v = 0; v < 8; ++v)
        pmax[r][v] = fmaxf(pmax[r][v], __shfl_xor(pmax[r][v], off, 32));
  if (l15 == 0) {
#pragma unroll
    for (int r = 0; r < 4; ++r)
#pragma unroll
      for (int v = 0; v < 8; ++v)
        redbuf[wave][(r << 4) + (hl << 3) + v] = pmax[r][v];
  }
  __syncthreads();
  if (threadIdx.x < 64) {
    float m = redbuf[0][threadIdx.x];
    for (int w = 1; w < 8; ++w) m = fmaxf(m, redbuf[w][threadIdx.x]);
    rowstat[threadIdx.x] = m;
  }
  __syncthreads();

  // --- exp + row sum ---
  float rmax[4][8], psum[4][8];
#pragma unroll
  for (int r = 0; r < 4; ++r)
#pragma unroll
    for (int v = 0; v < 8; ++v) {
      rmax[r][v] = rowstat[(r << 4) + (hl << 3) + v];
      psum[r][v] = 0.f;
    }
#pragma unroll
  for (int r = 0; r < 4; ++r)
#pragma unroll
    for (int t = 0; t < 8; ++t)
#pragma unroll
      for (int v = 0; v < 8; ++v) {
        float e = __expf(acc[r][t][v] - rmax[r][v]);   // v_exp_f32
        acc[r][t][v] = e;
        psum[r][v] += e;
      }
  for (int off = 8; off; off >>= 1)
#pragma unroll
    for (int r = 0; r < 4; ++r)
#pragma unroll
      for (int v = 0; v < 8; ++v)
        psum[r][v] += __shfl_xor(psum[r][v], off, 32);
  __syncthreads();                 // all rowstat(max) reads done
  if (l15 == 0) {
#pragma unroll
    for (int r = 0; r < 4; ++r)
#pragma unroll
      for (int v = 0; v < 8; ++v)
        redbuf[wave][(r << 4) + (hl << 3) + v] = psum[r][v];
  }
  __syncthreads();
  if (threadIdx.x < 64) {
    float s = redbuf[0][threadIdx.x];
    for (int w = 1; w < 8; ++w) s += redbuf[w][threadIdx.x];
    rowstat[threadIdx.x] = s;
  }
  __syncthreads();

  float rinv[4][8];
#pragma unroll
  for (int r = 0; r < 4; ++r)
#pragma unroll
    for (int v = 0; v < 8; ++v)
      rinv[r][v] = INV_KEEP / rowstat[(r << 4) + (hl << 3) + v];

  // --- dropout + transposed bf16 store ---
  // Lane owns k = wk0+16t+l15 with contiguous j = j0+r*16+8*hl.. -> 16B store.
#pragma unroll
  for (int r = 0; r < 4; ++r)
    for (int t = 0; t < 8; ++t) {
      const int k = wk0 + (t << 4) + l15;
      union { unsigned short s[8]; uint4 u; } P;
#pragma unroll
      for (int v = 0; v < 8; ++v) {
        const int m = (r << 4) + (hl << 3) + v;
        unsigned int idx = ((((unsigned)b << 10) | (unsigned)(j0 + m)) << 10)
                           | (unsigned)k;      // flat index into [B,S,S]
        float val = acc[r][t][v] * rinv[r][v];
        if (hash_u01(idx) >= KEEP_P) val = 0.0f;
        P.s[v] = f32_to_bf16(val);
      }
      *(uint4*)(attnT + (size_t)(b * S + k) * S + j0 + (r << 4) + (hl << 3)) =
          P.u;
    }
}

// ---------------- GEMM2: out[b][i][k] = value_w[i][:] . attnT[b][k][:] -----
// grid: (64, B) -> blockIdx.x = {i-block 0..7} x {k-block 0..7}, 128x128 tile.
// 8 waves each own a 16x128 strip; B tiles shared across waves via WGP$.
__global__ void __launch_bounds__(256) av_gemm(
    const unsigned short* __restrict__ vb,     // value_w bf16 [D,D]
    const unsigned short* __restrict__ attnT,  // [B,S,S] bf16, [b][k][j]
    float* __restrict__ out) {                 // [B,D,S] fp32
  const int wave = threadIdx.x >> 5;
  const int lane = threadIdx.x & 31;
  const int hl   = lane >> 4;
  const int l15  = lane & 15;
  const int b    = blockIdx.y;
  const int i0   = ((blockIdx.x >> 3) << 7) + (wave << 4);  // 128-row i-block
  const int k0   = (blockIdx.x & 7) << 7;                   // 128-col k-block

  const v8f vzero = {0.f,0.f,0.f,0.f,0.f,0.f,0.f,0.f};
  v8f acc[8];
#pragma unroll
  for (int t = 0; t < 8; ++t) acc[t] = vzero;

  const uint4* Abase = (const uint4*)(vb + (size_t)(i0 + l15) * D);

  for (int j = 0; j < S; j += 32) {
    const int c0 = (j >> 3) + hl;
    ABu A;
    A.u[0] = Abase[c0];
    A.u[1] = Abase[c0 + 2];
    __builtin_prefetch(Abase + c0 + 4, 0, 1);
#pragma unroll
    for (int t = 0; t < 8; ++t) {
      // B[K=j.., N=k]: lane's column k = k0+16t+l15 is a contiguous row of
      // attnT -> two b128 loads; identical across the 8 waves (WGP$ hits).
      const uint4* Bp = (const uint4*)(attnT
          + (size_t)(b * S + k0 + (t << 4) + l15) * S + j + (hl << 3));
      ABu Bt;
      Bt.u[0] = Bp[0];
      Bt.u[1] = Bp[2];
      acc[t] = __builtin_amdgcn_wmma_f32_16x16x32_bf16(
          false, A.v, false, Bt.v, (short)0, acc[t], false, false);
    }
  }

  // Coalesced fp32 stores: per (t,v) lanes 0-15 / 16-31 each write a
  // contiguous 64B row segment.
#pragma unroll
  for (int t = 0; t < 8; ++t)
#pragma unroll
    for (int v = 0; v < 8; ++v)
      out[(size_t)(b * D + i0 + v + (hl << 3)) * S + k0 + (t << 4) + l15] =
          acc[t][v];
}

// ---------------------------------------------------------------------------
extern "C" void kernel_launch(void* const* d_in, const int* in_sizes, int n_in,
                              void* d_out, int out_size, void* d_ws, size_t ws_size,
                              hipStream_t stream) {
  (void)in_sizes; (void)n_in; (void)out_size; (void)ws_size;
  const float* x  = (const float*)d_in[0];   // [B,S,D]
  const float* kw = (const float*)d_in[1];   // [D,D]
  const float* vw = (const float*)d_in[2];   // [D,D]
  float* out = (float*)d_out;                // [B,D,S]

  // Workspace layout (bf16): 68 MiB total.
  unsigned short* xb    = (unsigned short*)d_ws;           // 32 MiB
  unsigned short* kb    = xb + (size_t)B * S * D;          //  2 MiB
  unsigned short* vb    = kb + (size_t)D * D;              //  2 MiB
  unsigned short* attnT = vb + (size_t)D * D;              // 32 MiB

  const int n4x = B * S * D / 4;   // 4M float4 groups
  const int n4w = D * D / 4;       // 256K
  hipLaunchKernelGGL(cvt_bf16, dim3((n4x + 255) / 256), dim3(256), 0, stream, x,  xb, n4x);
  hipLaunchKernelGGL(cvt_bf16, dim3((n4w + 255) / 256), dim3(256), 0, stream, kw, kb, n4w);
  hipLaunchKernelGGL(cvt_bf16, dim3((n4w + 255) / 256), dim3(256), 0, stream, vw, vb, n4w);

  hipLaunchKernelGGL(qk_softmax_dropout, dim3(B * S / 64), dim3(256), 0, stream,
                     xb, kb, attnT);
  hipLaunchKernelGGL(av_gemm, dim3(64, B), dim3(256), 0, stream,
                     vb, attnT, out);
}
